// MyModel_57088705299066
// MI455X (gfx1250) — matrix operand order repaired and checked by reading, not compiled, not run on hardware
//
#include <hip/hip_runtime.h>
#include <math.h>

#define BATCH 64
#define SEQ   512
#define EMBD  128
#define HID   256
#define H4    1024
#define TAGS  9
#define MTOT  (BATCH*SEQ)          // 32768
#define LOGITS_N (MTOT*TAGS)       // 294912
#define HSTR  264                  // bf16 LDS row stride (528B, 16B aligned)

typedef __attribute__((ext_vector_type(2)))  float  v2f;
typedef __attribute__((ext_vector_type(4)))  float  v4f;
typedef __attribute__((ext_vector_type(8)))  float  v8f;
typedef __bf16 bf16_t;
typedef __attribute__((ext_vector_type(8)))  __bf16 v8bf;
typedef __attribute__((ext_vector_type(16))) __bf16 v16bf;

__device__ __forceinline__ v8f wmma_bf(v16bf a, v16bf b, v8f c) {
  // V_WMMA_F32_16X16X32_BF16 : D = A(16x32) * B(32x16) + C(16x16 f32)
  return __builtin_amdgcn_wmma_f32_16x16x32_bf16(false, a, false, b, (short)0, c,
                                                 false, false);
}
__device__ __forceinline__ v8f wmma4(v2f a, v2f b, v8f c) {
  // V_WMMA_F32_16X16X4_F32 (full fp32 path, decoder GEMM)
  return __builtin_amdgcn_wmma_f32_16x16x4_f32(false, a, false, b, (short)0, c,
                                               false, false);
}
__device__ __forceinline__ v16bf cat8(v8bf lo, v8bf hi) {
  return __builtin_shufflevector(lo, hi, 0, 1, 2, 3, 4, 5, 6, 7, 8, 9, 10, 11,
                                 12, 13, 14, 15);
}
__device__ __forceinline__ float sigmoidf_(float x) {
  return 1.f / (1.f + __expf(-x));
}
__device__ __forceinline__ float tanhf_(float x) {
  x = fminf(fmaxf(x, -15.f), 15.f);
  float e = __expf(2.f * x);
  return (e - 1.f) / (e + 1.f);
}

// ---------------------------------------------------------------- lens
__global__ void __launch_bounds__(256) k_lens(const int* __restrict__ text,
                                              int* __restrict__ lens) {
  __shared__ int red[256];
  int b = blockIdx.x;
  int cnt = 0;
  for (int s = threadIdx.x; s < SEQ; s += 256) cnt += (text[b * SEQ + s] != 0);
  red[threadIdx.x] = cnt;
  __syncthreads();
  for (int off = 128; off > 0; off >>= 1) {
    if ((int)threadIdx.x < off) red[threadIdx.x] += red[threadIdx.x + off];
    __syncthreads();
  }
  if (threadIdx.x == 0) lens[b] = red[0];
}

// ------------------------------------------------ pack U into B-fragment order
// UPK[dir][w:4][kk:3][g:2][lane:5][j:4] = bf16( U[kk*32 + (j<8 ? half*8 + j : 16 + half*8 + j-8)][g*256 + w*16 + ln] )
__global__ void __launch_bounds__(256) k_pack_u(const float* __restrict__ Uf,
                                                const float* __restrict__ Ub,
                                                bf16_t* __restrict__ upk) {
  const int o = blockIdx.x * 256 + threadIdx.x;  // 0..262143
  const int dir = blockIdx.y;
  const float* U = dir ? Ub : Uf;
  int j = o & 15, lane = (o >> 4) & 31, g = (o >> 9) & 3, kk = (o >> 11) & 7,
      w = (o >> 14) & 15;
  int half = lane >> 4, ln = lane & 15;
  int K = kk * 32 + ((j < 8) ? half * 8 + j : 16 + half * 8 + (j - 8));
  int col = g * 256 + w * 16 + ln;
  upk[(size_t)dir * 262144 + o] = (bf16_t)U[(size_t)K * H4 + col];
}

// WPK[dir][ntile:6][kk:2][lane:5][j:4] = bf16( W[kk*32 + (j<8 ? half*8 + j : 16 + half*8 + j-8)][ntile*16 + ln] )
__global__ void __launch_bounds__(256) k_pack_w(const float* __restrict__ Wf,
                                                const float* __restrict__ Wb,
                                                bf16_t* __restrict__ wpk) {
  const int o = blockIdx.x * 256 + threadIdx.x;  // 0..131071
  const int dir = blockIdx.y;
  const float* W = dir ? Wb : Wf;
  int j = o & 15, lane = (o >> 4) & 31, kk = (o >> 9) & 3, ntile = (o >> 11);
  int half = lane >> 4, ln = lane & 15;
  int K = kk * 32 + ((j < 8) ? half * 8 + j : 16 + half * 8 + (j - 8));
  int col = ntile * 16 + ln;
  wpk[(size_t)dir * 131072 + o] = (bf16_t)W[(size_t)K * H4 + col];
}

// ---------------------- xz = emb[text] @ W + b, written in LSTM packed layout
// XZP[(dir*4+q)*512 + s][w:4][lane:5][g:2][i:3] (f32)
__global__ void __launch_bounds__(256) k_xz(const int* __restrict__ text,
                                            const float* __restrict__ emb,
                                            const bf16_t* __restrict__ wpk,
                                            const float* __restrict__ bf,
                                            const float* __restrict__ bb,
                                            float* __restrict__ xzp) {
  __shared__ bf16_t As[16 * 136];  // 16 rows x 128 K (bf16), padded stride
  const int mtile = blockIdx.x;
  const int dir = blockIdx.y;
  const bf16_t* wp = wpk + (size_t)dir * 131072;
  const float* bias = dir ? bb : bf;

  for (int idx = threadIdx.x; idx < 16 * 128; idx += 256) {
    int r = idx >> 7, c = idx & 127;
    int tok = text[mtile * 16 + r];
    As[r * 136 + c] = (bf16_t)emb[(size_t)tok * EMBD + c];
  }
  __syncthreads();

  const int wid = threadIdx.x >> 5;
  const int lane = threadIdx.x & 31;
  const int half = lane >> 4;
  const int ln = lane & 15;

  // fixed per mtile: which (batch, quarter, fragment slot) this maps to
  const int m0 = mtile * 16;
  const int s0 = m0 & 511;
  const int bq = m0 >> 9;        // global batch row
  const int q = (bq >> 4) & 3;   // batch quarter
  const int r = bq & 15;         // row within quarter
  const int lane_l = ((r & 8) ? 16 : 0) + ln;
  const int i_l = r & 7;
  const size_t sbase = (size_t)(dir * 4 + q) * 512 * 16384;

  const bf16_t* pr = As + ln * 136;
  for (int j = 0; j < 8; j++) {
    int ntile = wid * 8 + j;
    int col = ntile * 16 + ln;
    v8f acc = {};
#pragma unroll
    for (int kk = 0; kk < 4; kk++) {
      int kb = kk * 32;
      v8bf alo = *(const v8bf*)(pr + kb + half * 8);
      v8bf ahi = *(const v8bf*)(pr + kb + 16 + half * 8);
      const bf16_t* bp = wp + ((size_t)(ntile * 4 + kk) * 32 + lane) * 16;
      v8bf blo = *(const v8bf*)(bp);
      v8bf bhi = *(const v8bf*)(bp + 8);
      acc = wmma_bf(cat8(alo, ahi), cat8(blo, bhi), acc);
    }
    float bv = bias[col];
    int g = ntile >> 4, wl = ntile & 15;
    size_t eb = sbase + (size_t)(wl * 32 + lane_l) * 32 + g * 8 + i_l;
#pragma unroll
    for (int i = 0; i < 8; i++) {
      int row = half ? i + 8 : i;
      xzp[eb + (size_t)(s0 + row) * 16384] = acc[i] + bv;
    }
  }
}

// --------------------------------------------------------------- LSTM scan
// grid (4 quarters, 2 dirs), 512 thr = 16 waves; wave w owns cols [16w,16w+16)
// of all 4 gates; c-state in registers; h exchanged via bf16 LDS buffer.
__global__ void __launch_bounds__(512) k_lstm(const bf16_t* __restrict__ upk,
                                              const float* __restrict__ xzp,
                                              float* __restrict__ hfo,
                                              float* __restrict__ hbo) {
  const int dir = blockIdx.y;
  const int q = blockIdx.x;
  const int b0 = q * 16;
  const bf16_t* up = upk + (size_t)dir * 262144;
  float* hout = dir ? hbo : hfo;

  __shared__ bf16_t hbuf[16 * HSTR];
  const int tid = threadIdx.x;
  const int w = tid >> 5, lane = tid & 31, half = lane >> 4, ln = lane & 15;
  const int mycol = w * 16 + ln;

  for (int i = tid; i < 16 * HSTR; i += 512) hbuf[i] = (bf16_t)0.f;
  v8f cst = {};
  __syncthreads();

  const bf16_t* pr = hbuf + ln * HSTR;
  const size_t lane_off = (size_t)(w * 32 + lane) * 32;
  const size_t dqb = (size_t)(dir * 4 + q) * 512;

  for (int t = 0; t < SEQ; t++) {
    int tt = dir ? (SEQ - 1 - t) : t;
    // gates start at packed xz (8x b128 contiguous per lane)
    const v4f* px = (const v4f*)(xzp + (dqb + tt) * 16384 + lane_off);
    v8f acc[4];
#pragma unroll
    for (int g = 0; g < 4; g++) {
      v4f c0 = px[g * 2 + 0];
      v4f c1 = px[g * 2 + 1];
      acc[g] = __builtin_shufflevector(c0, c1, 0, 1, 2, 3, 4, 5, 6, 7);
    }
    // += h @ U : 8 bf16 WMMAs per gate tile (K=256)
#pragma unroll 2
    for (int kk = 0; kk < 8; kk++) {
      int kb = kk * 32;
      v8bf alo = *(const v8bf*)(pr + kb + half * 8);
      v8bf ahi = *(const v8bf*)(pr + kb + 16 + half * 8);
      v16bf a = cat8(alo, ahi);
      const bf16_t* ub = up + (size_t)(w * 8 + kk) * 2048 + lane * 16;
#pragma unroll
      for (int g = 0; g < 4; g++) {
        const bf16_t* ug = ub + g * 512;
        v8bf blo = *(const v8bf*)(ug);
        v8bf bhi = *(const v8bf*)(ug + 8);
        acc[g] = wmma_bf(a, cat8(blo, bhi), acc[g]);
      }
    }
    // elementwise gate update (c-state wave-local, f32)
    float hv[8];
#pragma unroll
    for (int i = 0; i < 8; i++) {
      float ig = sigmoidf_(acc[0][i]);
      float fg = sigmoidf_(acc[1][i]);
      float gg = tanhf_(acc[2][i]);
      float og = sigmoidf_(acc[3][i]);
      float c = fg * cst[i] + ig * gg;
      cst[i] = c;
      hv[i] = og * tanhf_(c);
    }
    __syncthreads();  // all waves done reading hbuf(t)
#pragma unroll
    for (int i = 0; i < 8; i++) {
      int row = half ? i + 8 : i;
      hbuf[row * HSTR + mycol] = (bf16_t)hv[i];
      hout[((size_t)(b0 + row) * SEQ + tt) * HID + mycol] = hv[i];
    }
    __syncthreads();  // hbuf(t+1) visible
  }
}

// ----------------------------------------------- logits = [hf|hb] @ Wd + bd
// fp32 WMMA path; one wave per 16-row M-tile; N=9 padded to one 16-wide tile
__global__ void __launch_bounds__(256) k_logits(const float* __restrict__ hf,
                                                const float* __restrict__ hb,
                                                const float* __restrict__ Wd,
                                                const float* __restrict__ bd,
                                                float* __restrict__ out) {
  const int wid = threadIdx.x >> 5, lane = threadIdx.x & 31;
  const int half = lane >> 4, ln = lane & 15;
  const int m0 = (blockIdx.x * 8 + wid) * 16;
  v8f acc = {};
  for (int kk = 0; kk < 128; kk++) {
    int kb = kk * 4;
    const float* hsrc = (kb < 256) ? hf : hb;
    int kl = (kb < 256) ? kb : kb - 256;
    int k0 = kl + (half ? 2 : 0);
    int k1 = kl + (half ? 3 : 1);
    v2f a, b;
    a[0] = hsrc[(size_t)(m0 + ln) * HID + k0];
    a[1] = hsrc[(size_t)(m0 + ln) * HID + k1];
    int kg0 = kb + (half ? 2 : 0);
    int kg1 = kb + (half ? 3 : 1);
    b[0] = (ln < TAGS) ? Wd[kg0 * TAGS + ln] : 0.f;
    b[1] = (ln < TAGS) ? Wd[kg1 * TAGS + ln] : 0.f;
    acc = wmma4(a, b, acc);
  }
  if (ln < TAGS) {
    float bv = bd[ln];
#pragma unroll
    for (int i = 0; i < 8; i++) {
      int row = half ? i + 8 : i;
      out[(size_t)(m0 + row) * TAGS + ln] = acc[i] + bv;
    }
  }
}

// --------------------------------------------------------------- CRF
__global__ void __launch_bounds__(32) k_crf(const float* __restrict__ logits,
                                            const int* __restrict__ labels,
                                            const int* __restrict__ lens,
                                            const float* __restrict__ trans,
                                            float* __restrict__ out_lens,
                                            float* __restrict__ out_ll) {
  __shared__ float str[TAGS * TAGS];
  __shared__ float salpha[16];
  const int b = blockIdx.x;
  const int l = threadIdx.x;
  const int len = lens[b];
  for (int i = l; i < TAGS * TAGS; i += 32) str[i] = trans[i];
  __syncthreads();

  const float* lg = logits + (size_t)b * SEQ * TAGS;
  const int* lab = labels + b * SEQ;

  float part = 0.f;
  for (int t = l; t < SEQ; t += 32) {
    if (t < len) part += lg[t * TAGS + lab[t]];
    if (t < len - 1) part += str[lab[t] * TAGS + lab[t + 1]];
  }
  for (int off = 16; off; off >>= 1) part += __shfl_xor(part, off, 32);
  const float score = part;

  if (l < TAGS) salpha[l] = lg[l];
  __syncthreads();
  for (int t = 1; t < SEQ; t++) {
    float nv = 0.f;
    if (l < TAGS) {
      float mx = -1e30f;
#pragma unroll
      for (int i = 0; i < TAGS; i++) mx = fmaxf(mx, salpha[i] + str[i * TAGS + l]);
      float s = 0.f;
#pragma unroll
      for (int i = 0; i < TAGS; i++) s += __expf(salpha[i] + str[i * TAGS + l] - mx);
      float newv = mx + __logf(s) + lg[t * TAGS + l];
      nv = (t < len) ? newv : salpha[l];
    }
    __syncthreads();
    if (l < TAGS) salpha[l] = nv;
    __syncthreads();
  }
  if (l == 0) {
    float mx = -1e30f;
    for (int i = 0; i < TAGS; i++) mx = fmaxf(mx, salpha[i]);
    float s = 0.f;
    for (int i = 0; i < TAGS; i++) s += __expf(salpha[i] - mx);
    out_ll[b] = score - (mx + __logf(s));
    out_lens[b] = (float)len;
  }
}

extern "C" void kernel_launch(void* const* d_in, const int* in_sizes, int n_in,
                              void* d_out, int out_size, void* d_ws,
                              size_t ws_size, hipStream_t stream) {
  (void)in_sizes; (void)n_in; (void)out_size; (void)ws_size;
  const int*   text   = (const int*)d_in[0];
  const int*   labels = (const int*)d_in[1];
  const float* emb    = (const float*)d_in[2];
  const float* Wf     = (const float*)d_in[3];
  const float* Uf     = (const float*)d_in[4];
  const float* bf     = (const float*)d_in[5];
  const float* Wb     = (const float*)d_in[6];
  const float* Ub     = (const float*)d_in[7];
  const float* bb     = (const float*)d_in[8];
  const float* Wd     = (const float*)d_in[9];
  const float* bd     = (const float*)d_in[10];
  const float* trans  = (const float*)d_in[11];
  float* out = (float*)d_out;

  float* ws  = (float*)d_ws;
  float*  xzp = ws;                                   // 67,108,864 f32 (packed, both dirs)
  float*  hf  = xzp + (size_t)2 * MTOT * H4;          //  8,388,608 f32
  float*  hb  = hf + (size_t)MTOT * HID;              //  8,388,608 f32
  bf16_t* upk = (bf16_t*)(hb + (size_t)MTOT * HID);   //  2 x 262,144 bf16
  bf16_t* wpk = upk + 2 * 262144;                     //  2 x 131,072 bf16
  int*    lens = (int*)(wpk + 2 * 131072);            //  64 i32

  k_lens<<<BATCH, 256, 0, stream>>>(text, lens);
  k_pack_u<<<dim3(1024, 2), 256, 0, stream>>>(Uf, Ub, upk);
  k_pack_w<<<dim3(512, 2), 256, 0, stream>>>(Wf, Wb, wpk);
  k_xz<<<dim3(MTOT / 16, 2), 256, 0, stream>>>(text, emb, wpk, bf, bb, xzp);
  k_lstm<<<dim3(4, 2), 512, 0, stream>>>(upk, xzp, hf, hb);
  k_logits<<<MTOT / 16 / 8, 256, 0, stream>>>(hf, hb, Wd, bd, out);
  k_crf<<<BATCH, 32, 0, stream>>>(out, labels, lens, trans,
                                  out + LOGITS_N, out + LOGITS_N + BATCH);
}
